// MultiHeadAttention_LaTeX_36429912605014
// MI455X (gfx1250) — compile-verified, matrix operand order checked by reading, not run on hardware
//
#include <hip/hip_runtime.h>
#include <hip/hip_bf16.h>

// ---------------------------------------------------------------------------
// MI455X (gfx1250) fused MHA+RoPE, wave32, v_wmma_f32_16x16x32_f16 everywhere.
// Flash-attention fusion avoids materializing the 36 GB score tensor; f16
// operand "panels" (lane-contiguous 32B chunks) make every WMMA operand a
// single coalesced global/LDS load. x tiles staged via async-to-LDS (ASYNCcnt),
// cross-lane ops via ds_swizzle xor patterns, RoPE tables staged in LDS.
// ---------------------------------------------------------------------------

typedef __attribute__((ext_vector_type(16))) _Float16 v16h;
typedef __attribute__((ext_vector_type(8)))  float    v8f;

constexpr int B = 128, T = 189, C = 128, H = 32, D = 128;
constexpr int TPAD = 192;           // padded sequence length
constexpr int TT16 = 12;            // 16-row tiles over TPAD
constexpr int TT32 = 6;             // 32-row tiles
constexpr float QSCALE = 0.08838834764831845f;   // 1/sqrt(C)

// workspace layout (bytes)
constexpr size_t OFF_WQ  = 0;
constexpr size_t OFF_WK  = (size_t)1 << 20;
constexpr size_t OFF_WV  = (size_t)2 << 20;
constexpr size_t OFF_WP  = (size_t)3 << 20;
constexpr size_t OFF_COS = (size_t)4 << 20;
constexpr size_t OFF_SIN = OFF_COS + (1 << 16);
constexpr size_t OFF_QP  = (size_t)8 << 20;
constexpr size_t SZ_P    = (size_t)B * H * TT16 * 4 * 512 * 2;   // 192 MB
constexpr size_t OFF_KP  = OFF_QP + SZ_P;
constexpr size_t OFF_VP  = OFF_KP + SZ_P;
constexpr size_t OFF_OP  = OFF_VP + SZ_P;

__device__ __forceinline__ v8f wmma_f16(v16h a, v16h b, v8f c) {
  return __builtin_amdgcn_wmma_f32_16x16x32_f16(false, a, false, b,
                                                (short)0, c, false, false);
}
// ISA A/B-operand K index for f16 16x16x32 (CDNA5 ISA 7.12.2)
__device__ __forceinline__ int kmap(int hh, int grp) {
  int v = hh >> 1;
  return ((v < 4) ? 0 : 16) + 8 * grp + 2 * (v & 3) + (hh & 1);
}
__device__ __forceinline__ void lds_fence() {
  asm volatile("s_wait_dscnt 0" ::: "memory");
  __builtin_amdgcn_wave_barrier();
}
// xor-butterfly via ds_swizzle (group-of-32: and=0x1f, or=0, xor=XM)
template <int XM>
__device__ __forceinline__ float swz(float v) {
  return __int_as_float(
      __builtin_amdgcn_ds_swizzle(__float_as_int(v), (XM << 10) | 0x1f));
}

// ---------------- kernel 0a: pack Wq/Wk/Wv into B-operand panels -----------
// dst layout: (((h*4+ks)*8+dt)*512 + lane*16 + hh) halves
__global__ void pack_w_kernel(const float* __restrict__ W, _Float16* __restrict__ dst) {
  int i = blockIdx.x * 256 + threadIdx.x;          // exact multiple, no guard
  int h  = i / (4 * 8 * 512);
  int r  = i % (4 * 8 * 512);
  int ks = r / (8 * 512);
  int r2 = r % (8 * 512);
  int dt = r2 / 512;
  int lane = (r2 >> 4) & 31;
  int hh = r2 & 15;
  int c = ks * 32 + kmap(hh, lane >> 4);
  int d = dt * 16 + (lane & 15);
  dst[i] = (_Float16)W[((size_t)h * C + c) * D + d];
}

// ---------------- kernel 0b: pack Wp^T into B-operand panels ---------------
// dst layout: ((ks*8+dt)*512 + lane*16 + hh), element = Wp[col][k]
__global__ void pack_wp_kernel(const float* __restrict__ Wp, _Float16* __restrict__ dst) {
  int i = blockIdx.x * 256 + threadIdx.x;
  int ks = i / (8 * 512);
  int r2 = i % (8 * 512);
  int dt = r2 / 512;
  int lane = (r2 >> 4) & 31;
  int hh = r2 & 15;
  int k   = ks * 32 + kmap(hh, lane >> 4);
  int col = dt * 16 + (lane & 15);
  dst[i] = (_Float16)Wp[(size_t)col * (H * D) + k];
}

// ---------------- kernel 0c: RoPE tables (padded to TPAD rows) -------------
__global__ void rope_tables_kernel(float* __restrict__ cosT, float* __restrict__ sinT) {
  int i = blockIdx.x * 256 + threadIdx.x;
  if (i >= TPAD * 64) return;
  int t = i >> 6, j = i & 63;
  float inv = powf(10000.f, -(float)j / 32.f);    // theta^(-2*pairs/D), pairs=2j
  float ang = (float)t * inv;
  cosT[i] = cosf(ang);
  sinT[i] = sinf(ang);
}

// ---------------- kernel 1: QKV projection + RoPE, write panels ------------
// grid.x = B*TT32, block = 128 (4 waves); wave handles heads wid, wid+4, ...
__global__ void qkv_rope_kernel(const float* __restrict__ x,
                                const _Float16* __restrict__ wqp,
                                const _Float16* __restrict__ wkp,
                                const _Float16* __restrict__ wvp,
                                const float* __restrict__ cosT,
                                const float* __restrict__ sinT,
                                _Float16* __restrict__ qp,
                                _Float16* __restrict__ kp,
                                _Float16* __restrict__ vp) {
  // xraw is dead after panel packing; reused as the interleaved cos/sin tile.
  __shared__ float    xraw[32 * 128];      // 16KB: raw x tile / {cos,sin} tile
  __shared__ _Float16 xpan[8][512];        // A panels: [mt*4+ks][lane*16+hh]
  __shared__ _Float16 stage[4][32 * 128];  // per-wave f16 staging tile
  int b = blockIdx.x / TT32, tblk = blockIdx.x % TT32;
  int tid = threadIdx.x;

  // ---- bulk-copy x tile [32 rows,128 c] global->LDS via async DMA path ----
  {
    int row = tid >> 2;                    // 0..31
    int seg = tid & 3;                     // 128B segment within row
    int tg = tblk * 32 + row;
    int tsrc = (tg < T) ? tg : (T - 1);    // clamp: keep global reads in-bounds
    unsigned long long gaddr =
        (unsigned long long)(const char*)&x[((size_t)b * T + tsrc) * C] + seg * 128;
    unsigned laddr = (unsigned)(unsigned long long)&xraw[row * 128 + seg * 32];
#pragma unroll
    for (int kk = 0; kk < 8; ++kk) {       // 8 x 16B per thread
      asm volatile("global_load_async_to_lds_b128 %0, %1, off"
                   :: "v"(laddr), "v"(gaddr) : "memory");
      laddr += 16;
      gaddr += 16;
    }
    asm volatile("s_wait_asynccnt 0x0" ::: "memory");
  }
  __syncthreads();

  // ---- scatter-pack LDS raw tile into WMMA A-operand panels ----
  for (int i = tid; i < 8 * 512; i += 128) {
    int p = i >> 9, mt = p >> 2, ks = p & 3;
    int lane = (i >> 4) & 31, hh = i & 15;
    int c = ks * 32 + kmap(hh, lane >> 4);
    int row = mt * 16 + (lane & 15);
    int t = tblk * 32 + row;
    float val = (t < T) ? xraw[row * 128 + c] : 0.f;
    xpan[p][i & 511] = (_Float16)val;
  }
  __syncthreads();

  // ---- overlay xraw with interleaved {cos,sin} for this 32-row block ------
  float* csS = xraw;                       // [row][j][2] = 32*64*2 floats
  for (int i = tid; i < 32 * 64; i += 128) {
    int row = i >> 6, j = i & 63;
    int t = tblk * 32 + row;               // t < TPAD, tables padded
    csS[2 * i]     = cosT[t * 64 + j];
    csS[2 * i + 1] = sinT[t * 64 + j];
  }
  __syncthreads();

  int wid = tid >> 5, lane = tid & 31;
  int grp = lane >> 4, n = lane & 15;
  for (int h = wid; h < H; h += 4) {
    for (int m3 = 0; m3 < 3; ++m3) {
      const _Float16* wpan = (m3 == 0) ? wqp : (m3 == 1) ? wkp : wvp;
#pragma unroll 1
      for (int mt = 0; mt < 2; ++mt) {
#pragma unroll 1
        for (int dt = 0; dt < 8; ++dt) {
          v8f acc = {};
#pragma unroll
          for (int ks = 0; ks < 4; ++ks) {
            v16h a  = *(const v16h*)&xpan[mt * 4 + ks][lane * 16];
            v16h bm = *(const v16h*)&wpan[((((size_t)h * 4 + ks) * 8 + dt) * 512) + lane * 16];
            acc = wmma_f16(a, bm, acc);
          }
          if (m3 < 2) {                      // RoPE via lane-xor pairing
            int j = dt * 8 + (n >> 1);
            bool even = !(n & 1);
            int rowb = mt * 16 + 8 * grp;    // LDS ds_load_b64, imm offsets
#pragma unroll
            for (int r = 0; r < 8; ++r) {
              float2 cs = *(const float2*)&csS[((rowb + r) * 64 + j) * 2];
              float vo = acc[r];
              float vpart = swz<1>(vo);
              float o = even ? (vo * cs.x - vpart * cs.y) : (vpart * cs.y + vo * cs.x);
              if (m3 == 0) o *= QSCALE;      // fold 1/sqrt(C) into Q
              acc[r] = o;
            }
          }
#pragma unroll
          for (int r = 0; r < 8; ++r) {
            int row = mt * 16 + r + 8 * grp;
            stage[wid][row * 128 + dt * 16 + n] = (_Float16)acc[r];
          }
        }
      }
      lds_fence();
      if (m3 < 2) {                          // Q/K: two 16-row A/B panels x 4 ksteps
        _Float16* dst = (m3 == 0) ? qp : kp;
        for (int mt = 0; mt < 2; ++mt)
          for (int ks = 0; ks < 4; ++ks) {
            v16h pk;
#pragma unroll
            for (int hh = 0; hh < 16; ++hh)
              pk[hh] = stage[wid][(mt * 16 + n) * 128 + ks * 32 + kmap(hh, grp)];
            size_t off = ((((size_t)b * H + h) * TT16 + (tblk * 2 + mt)) * 4 + ks) * 512 + lane * 16;
            *(v16h*)&dst[off] = pk;
          }
      } else {                               // V: 8 d-tile B panels over 32 key rows
        for (int dt = 0; dt < 8; ++dt) {
          v16h pk;
#pragma unroll
          for (int hh = 0; hh < 16; ++hh)
            pk[hh] = stage[wid][kmap(hh, grp) * 128 + dt * 16 + n];
          size_t off = ((((size_t)b * H + h) * TT32 + tblk) * 8 + dt) * 512 + lane * 16;
          *(v16h*)&vp[off] = pk;
        }
      }
      lds_fence();
    }
  }
}

// ---------------- kernel 2: flash attention (one wave per 16-row q-tile) ---
// grid.x = B*H*3, block = 128 (4 waves)
__global__ void flash_attn_kernel(const _Float16* __restrict__ qp,
                                  const _Float16* __restrict__ kp,
                                  const _Float16* __restrict__ vp,
                                  const int* __restrict__ mask,
                                  _Float16* __restrict__ op) {
  __shared__ _Float16 pbuf[4][512];       // per-wave P tile  [16][32]
  __shared__ _Float16 obuf[4][2048];      // per-wave O tile  [16][128]
  int wid = threadIdx.x >> 5, lane = threadIdx.x & 31;
  int gq = blockIdx.x * 4 + wid;
  int qt = gq % TT16; int bh = gq / TT16; int h = bh % H; int b = bh / H;
  int t0 = qt * 16;
  int grp = lane >> 4, n = lane & 15;

  v16h qA[4];
#pragma unroll
  for (int ks = 0; ks < 4; ++ks)
    qA[ks] = *(const v16h*)&qp[((((size_t)b * H + h) * TT16 + qt) * 4 + ks) * 512 + lane * 16];

  float mrow[8], lrow[8];
  v8f acc[8];
#pragma unroll
  for (int r = 0; r < 8; ++r) { mrow[r] = -1e30f; lrow[r] = 0.f; }
#pragma unroll
  for (int dt = 0; dt < 8; ++dt) acc[dt] = (v8f){};

  int kbmax = (t0 + 15) / 32;
#pragma unroll 1
  for (int kb = 0; kb <= kbmax; ++kb) {
    const _Float16* vbase = &vp[((((size_t)b * H + h) * TT32 + kb) * 8) * 512];
    __builtin_prefetch(vbase + (size_t)lane * 256, 0, 1);   // global_prefetch_b8
    v8f S0 = {}, S1 = {};
#pragma unroll
    for (int ks = 0; ks < 4; ++ks) {
      v16h k0 = *(const v16h*)&kp[((((size_t)b * H + h) * TT16 + kb * 2) * 4 + ks) * 512 + lane * 16];
      v16h k1 = *(const v16h*)&kp[((((size_t)b * H + h) * TT16 + kb * 2 + 1) * 4 + ks) * 512 + lane * 16];
      S0 = wmma_f16(qA[ks], k0, S0);
      S1 = wmma_f16(qA[ks], k1, S1);
    }
    int s0c = kb * 32 + n, s1c = s0c + 16;
    int mk0 = (s0c < T) ? mask[(size_t)b * T + s0c] : 0;
    int mk1 = (s1c < T) ? mask[(size_t)b * T + s1c] : 0;
#pragma unroll
    for (int r = 0; r < 8; ++r) {
      int t = t0 + r + 8 * grp;
      float a0 = (mk0 && s0c <= t) ? S0[r] : -1e30f;   // causal + latex mask
      float a1 = (mk1 && s1c <= t) ? S1[r] : -1e30f;
      float rm = fmaxf(a0, a1);
      rm = fmaxf(rm, swz<1>(rm));
      rm = fmaxf(rm, swz<2>(rm));
      rm = fmaxf(rm, swz<4>(rm));
      rm = fmaxf(rm, swz<8>(rm));
      float mnew = fmaxf(mrow[r], rm);
      float sc = __expf(mrow[r] - mnew);
      mrow[r] = mnew;
      float p0 = __expf(a0 - mnew), p1 = __expf(a1 - mnew);
      float rs = p0 + p1;
      rs += swz<1>(rs);
      rs += swz<2>(rs);
      rs += swz<4>(rs);
      rs += swz<8>(rs);
      lrow[r] = lrow[r] * sc + rs;
#pragma unroll
      for (int dt = 0; dt < 8; ++dt) acc[dt][r] *= sc;
      int row = r + 8 * grp;
      pbuf[wid][row * 32 + n]      = (_Float16)p0;
      pbuf[wid][row * 32 + 16 + n] = (_Float16)p1;
    }
    lds_fence();
    v16h pA;                                   // C-layout -> A-layout via LDS
#pragma unroll
    for (int hh = 0; hh < 16; ++hh)
      pA[hh] = pbuf[wid][n * 32 + kmap(hh, grp)];
#pragma unroll
    for (int dt = 0; dt < 8; ++dt) {
      v16h vB = *(const v16h*)&vbase[dt * 512 + lane * 16];
      acc[dt] = wmma_f16(pA, vB, acc[dt]);
    }
    lds_fence();
  }
  // normalize + emit O as A-operand panels for the output projection
#pragma unroll
  for (int r = 0; r < 8; ++r) {
    float inv = lrow[r] > 0.f ? 1.f / lrow[r] : 0.f;
    int t = t0 + r + 8 * grp;
    int row = r + 8 * grp;
#pragma unroll
    for (int dt = 0; dt < 8; ++dt) {
      float o = (t < T) ? acc[dt][r] * inv : 0.f;
      obuf[wid][row * 128 + dt * 16 + n] = (_Float16)o;
    }
  }
  lds_fence();
#pragma unroll
  for (int ks = 0; ks < 4; ++ks) {
    v16h pk;
#pragma unroll
    for (int hh = 0; hh < 16; ++hh)
      pk[hh] = obuf[wid][n * 128 + ks * 32 + kmap(hh, grp)];
    size_t off = (((size_t)b * TT16 + qt) * 128 + (h * 4 + ks)) * 512 + lane * 16;
    *(v16h*)&op[off] = pk;
  }
}

// ---------------- kernel 3: output projection (K=4096) + bias --------------
// grid.x = B*TT16, block = 128 (4 waves, 2 n-tiles each)
__global__ void out_proj_kernel(const _Float16* __restrict__ op,
                                const _Float16* __restrict__ wpp,
                                const float* __restrict__ bp,
                                float* __restrict__ out) {
  int wid = threadIdx.x >> 5, lane = threadIdx.x & 31;
  int mt = blockIdx.x; int b = mt / TT16; int qt = mt % TT16;
  int grp = lane >> 4, n = lane & 15;
  int nt0 = wid * 2, nt1 = nt0 + 1;
  v8f a0 = {}, a1 = {};
#pragma unroll 1
  for (int ks = 0; ks < 128; ++ks) {
    v16h A  = *(const v16h*)&op[((size_t)mt * 128 + ks) * 512 + lane * 16];
    v16h B0 = *(const v16h*)&wpp[((size_t)ks * 8 + nt0) * 512 + lane * 16];
    v16h B1 = *(const v16h*)&wpp[((size_t)ks * 8 + nt1) * 512 + lane * 16];
    a0 = wmma_f16(A, B0, a0);
    a1 = wmma_f16(A, B1, a1);
  }
  float bb0 = bp[nt0 * 16 + n], bb1 = bp[nt1 * 16 + n];
#pragma unroll
  for (int r = 0; r < 8; ++r) {
    int t = qt * 16 + r + 8 * grp;
    if (t < T) {
      size_t base = ((size_t)b * T + t) * C;
      out[base + nt0 * 16 + n] = a0[r] + bb0;
      out[base + nt1 * 16 + n] = a1[r] + bb1;
    }
  }
}

// ---------------------------------------------------------------------------
extern "C" void kernel_launch(void* const* d_in, const int* in_sizes, int n_in,
                              void* d_out, int out_size, void* d_ws, size_t ws_size,
                              hipStream_t stream) {
  const float* x    = (const float*)d_in[0];
  const int*   mask = (const int*)d_in[1];
  const float* Wq   = (const float*)d_in[2];
  const float* Wk   = (const float*)d_in[3];
  const float* Wv   = (const float*)d_in[4];
  const float* Wp   = (const float*)d_in[5];
  const float* bp   = (const float*)d_in[6];
  float* out = (float*)d_out;

  char* w = (char*)d_ws;
  _Float16* wqp  = (_Float16*)(w + OFF_WQ);
  _Float16* wkp  = (_Float16*)(w + OFF_WK);
  _Float16* wvp  = (_Float16*)(w + OFF_WV);
  _Float16* wpp  = (_Float16*)(w + OFF_WP);
  float*    cosT = (float*)(w + OFF_COS);
  float*    sinT = (float*)(w + OFF_SIN);
  _Float16* qp   = (_Float16*)(w + OFF_QP);
  _Float16* kp   = (_Float16*)(w + OFF_KP);
  _Float16* vp   = (_Float16*)(w + OFF_VP);
  _Float16* op   = (_Float16*)(w + OFF_OP);

  // prep: weight panels + RoPE tables (weights stay L2-resident afterwards)
  pack_w_kernel<<<H * 4 * 8 * 512 / 256, 256, 0, stream>>>(Wq, wqp);
  pack_w_kernel<<<H * 4 * 8 * 512 / 256, 256, 0, stream>>>(Wk, wkp);
  pack_w_kernel<<<H * 4 * 8 * 512 / 256, 256, 0, stream>>>(Wv, wvp);
  pack_wp_kernel<<<128 * 8 * 512 / 256, 256, 0, stream>>>(Wp, wpp);
  rope_tables_kernel<<<(TPAD * 64 + 255) / 256, 256, 0, stream>>>(cosT, sinT);

  qkv_rope_kernel<<<B * TT32, 128, 0, stream>>>(x, wqp, wkp, wvp, cosT, sinT, qp, kp, vp);
  flash_attn_kernel<<<B * H * 3, 128, 0, stream>>>(qp, kp, vp, mask, op);
  out_proj_kernel<<<B * TT16, 128, 0, stream>>>(op, wpp, bp, out);
}
